// GCNCLF_55825984913950
// MI455X (gfx1250) — matrix operand order repaired and checked by reading, not compiled, not run on hardware
//
#include <hip/hip_runtime.h>
#include <math.h>

// ---------------------------------------------------------------------------
// GCN classifier w/ hierarchical pooling for MI455X (gfx1250, wave32, WMMA).
// All heavy math through V_WMMA_F32_16X16X32_F16 (f16 inputs, f32 accum).
// LDS layouts: every fragment read is ds_load_b128, every staging write is
// ds_store_b64. Interior tiles take a fully unguarded scalar-uniform fast
// path; ragged edges fall back to guarded loads.
// ---------------------------------------------------------------------------

typedef __attribute__((ext_vector_type(4)))  _Float16 v4h;
typedef __attribute__((ext_vector_type(8)))  _Float16 v8h;
typedef __attribute__((ext_vector_type(16))) _Float16 v16h;
typedef __attribute__((ext_vector_type(8)))  float    v8f;

#define BATCH 16
#define LDH   40   // LDS row stride in halves: 80B = 20 dwords (bank-conflict free)

enum { EPI_NONE = 0, EPI_RELU = 1, EPI_ADDI = 2 };

// ---------------------------------------------------------------------------
// Batched GEMM: C[z] = op(A[z]) * op(B[z])  (+ epilogue), row-major f32 I/O.
//   TA: A stored [K,M] row-major, logical A[m][k] = A[k*M+m]
//   TB: B stored [N,K] row-major, logical B[k][n] = B[n*K+k]
//   strideB == 0 => weight shared across batch.
// Block: 128 threads = 4 waves. Block tile 64x64; wave w -> rows [16w,16w+16).
// LDS: As[m][k] 64x32 (stride 40), Bs stored TRANSPOSED as Bs[n][k] 64x32.
// Fragment layouts follow CDNA5 ISA 7.12.2 (wave32).
// ---------------------------------------------------------------------------
template <int TA, int TB>
__global__ void gemm_wmma_f16(const float* __restrict__ A,
                              const float* __restrict__ B,
                              float* __restrict__ C,
                              int M, int N, int K,
                              long long strideA, long long strideB, long long strideC,
                              int epi) {
    __shared__ __align__(16) _Float16 As[64 * LDH];  // [m][k]
    __shared__ __align__(16) _Float16 Bs[64 * LDH];  // [n][k]  (transposed tile)

    const int tid    = threadIdx.x;
    const int lane   = tid & 31;
    const int waveId = tid >> 5;

    const int m0 = blockIdx.y * 64;
    const int n0 = blockIdx.x * 64;

    const float* Ab = A + (size_t)blockIdx.z * strideA;
    const float* Bb = B + (size_t)blockIdx.z * strideB;
    float*       Cb = C + (size_t)blockIdx.z * strideC;

    v8f acc[4] = {};

    const bool fullM = (m0 + 63 < M);
    const bool fullN = (n0 + 63 < N);

    const int kTiles = (K + 31) >> 5;
    for (int kt = 0; kt < kTiles; ++kt) {
        const int k0 = kt << 5;
        const bool fullK = (k0 + 31 < K) || ((K & 31) == 0 && k0 + 32 == K);

        // ---- stage A tile 64x32: 512 chunks of 4 k-contiguous halves -------
        if (fullM && fullK) {                 // scalar-uniform fast path
            #pragma unroll
            for (int i = 0; i < 4; ++i) {
                int chunk = i * 128 + tid;    // 0..511
                int r  = chunk >> 3;          // m row 0..63
                int c4 = (chunk & 7) << 2;    // k offset 0,4,..,28
                int gm = m0 + r, gk = k0 + c4;
                v4h h;
                if (!TA) {                    // contiguous -> global_load_b128
                    const float* p = Ab + (size_t)gm * K + gk;
                    h[0] = (_Float16)p[0]; h[1] = (_Float16)p[1];
                    h[2] = (_Float16)p[2]; h[3] = (_Float16)p[3];
                } else {                      // strided gather, unguarded
                    #pragma unroll
                    for (int j = 0; j < 4; ++j)
                        h[j] = (_Float16)Ab[(size_t)(gk + j) * M + gm];
                }
                *(v4h*)(&As[r * LDH + c4]) = h;   // ds_store_b64
            }
        } else {                              // ragged edge: guarded
            #pragma unroll
            for (int i = 0; i < 4; ++i) {
                int chunk = i * 128 + tid;
                int r  = chunk >> 3;
                int c4 = (chunk & 7) << 2;
                int gm = m0 + r, gk = k0 + c4;
                v4h h;
                #pragma unroll
                for (int j = 0; j < 4; ++j) {
                    int kk = gk + j;
                    float v = 0.0f;
                    if (gm < M && kk < K)
                        v = TA ? Ab[(size_t)kk * M + gm] : Ab[(size_t)gm * K + kk];
                    h[j] = (_Float16)v;
                }
                *(v4h*)(&As[r * LDH + c4]) = h;
            }
        }
        // ---- stage B tile transposed: Bs[n][k], 4 k-contiguous halves ------
        if (fullN && fullK) {                 // scalar-uniform fast path
            #pragma unroll
            for (int i = 0; i < 4; ++i) {
                int chunk = i * 128 + tid;    // 0..511
                int n  = chunk >> 3;          // n row 0..63
                int r4 = (chunk & 7) << 2;    // k offset
                int gn = n0 + n, gk = k0 + r4;
                v4h h;
                if (TB) {                     // contiguous -> global_load_b128
                    const float* p = Bb + (size_t)gn * K + gk;
                    h[0] = (_Float16)p[0]; h[1] = (_Float16)p[1];
                    h[2] = (_Float16)p[2]; h[3] = (_Float16)p[3];
                } else {                      // strided gather, unguarded
                    #pragma unroll
                    for (int j = 0; j < 4; ++j)
                        h[j] = (_Float16)Bb[(size_t)(gk + j) * N + gn];
                }
                *(v4h*)(&Bs[n * LDH + r4]) = h;   // ds_store_b64
            }
        } else {                              // ragged edge: guarded
            #pragma unroll
            for (int i = 0; i < 4; ++i) {
                int chunk = i * 128 + tid;
                int n  = chunk >> 3;
                int r4 = (chunk & 7) << 2;
                int gn = n0 + n, gk = k0 + r4;
                v4h h;
                #pragma unroll
                for (int j = 0; j < 4; ++j) {
                    int kk = gk + j;
                    float v = 0.0f;
                    if (gn < N && kk < K)
                        v = TB ? Bb[(size_t)gn * K + kk] : Bb[(size_t)kk * N + gn];
                    h[j] = (_Float16)v;
                }
                *(v4h*)(&Bs[n * LDH + r4]) = h;
            }
        }

        // hint the next K-tile into cache (global_prefetch_b8)
        if (kt + 1 < kTiles && fullM) {
            int r = tid >> 5, c = tid & 31;
            int gm = m0 + r, gk = k0 + 32 + c;
            if (gk < K) {
                const float* p = TA ? &Ab[(size_t)gk * M + gm]
                                    : &Ab[(size_t)gm * K + gk];
                __builtin_prefetch((const void*)p, 0, 3);
            }
        }
        __syncthreads();

        // ---- A fragment: lanes 0-15 K{0..7,16..23}, lanes 16-31 K{8..15,24..31}
        v16h af;
        {
            const int mr = waveId * 16 + (lane & 15);
            const int kh = (lane < 16) ? 0 : 8;
            v8h a0 = *(const v8h*)(&As[mr * LDH + kh]);        // ds_load_b128
            v8h a1 = *(const v8h*)(&As[mr * LDH + 16 + kh]);   // ds_load_b128
            af = __builtin_shufflevector(a0, a1, 0, 1, 2, 3, 4, 5, 6, 7,
                                                 8, 9, 10, 11, 12, 13, 14, 15);
        }
        // ---- 4 WMMAs across the 64-wide N tile, reusing af
        {
            const int nc = lane & 15;
            const int kb = (lane < 16) ? 0 : 8;
            #pragma unroll
            for (int t = 0; t < 4; ++t) {
                const int nr = t * 16 + nc;
                v8h b0 = *(const v8h*)(&Bs[nr * LDH + kb]);      // ds_load_b128
                v8h b1 = *(const v8h*)(&Bs[nr * LDH + kb + 16]); // ds_load_b128
                v16h bf = __builtin_shufflevector(b0, b1, 0, 1, 2, 3, 4, 5, 6, 7,
                                                          8, 9, 10, 11, 12, 13, 14, 15);
                acc[t] = __builtin_amdgcn_wmma_f32_16x16x32_f16(
                    /*neg_a=*/false, af, /*neg_b=*/false, bf,
                    /*c_mod=*/(short)0, acc[t],
                    /*reuse_a=*/false, /*reuse_b=*/false);
            }
        }
        __syncthreads();
    }

    // ---- epilogue + store (C layout: vgpr r -> M = (lane<16?0:8)+r, N = lane&15)
    #pragma unroll
    for (int t = 0; t < 4; ++t) {
        const int gn = n0 + t * 16 + (lane & 15);
        #pragma unroll
        for (int r = 0; r < 8; ++r) {
            const int gm = m0 + waveId * 16 + ((lane >> 4) * 8 + r);
            float v = acc[t][r];
            if (epi == EPI_RELU) v = fmaxf(v, 0.0f);
            if (epi == EPI_ADDI && gm == gn) v += 1.0f;
            if (gm < M && gn < N) Cb[(size_t)gm * N + gn] = v;
        }
    }
}

// ---------------------------------------------------------------------------
// dinv[b][i] = rsqrt(sum_j A[b][i][j]) ; one wave per row (wave32 reduction)
// grid: (ceil(N/4), B), block 128
// ---------------------------------------------------------------------------
__global__ void rowsum_rsqrt(const float* __restrict__ A, float* __restrict__ dinv, int N) {
    const int lane = threadIdx.x & 31;
    const int row  = blockIdx.x * (blockDim.x >> 5) + (threadIdx.x >> 5);
    const int b    = blockIdx.y;
    if (row >= N) return;
    const float* r = A + ((size_t)b * N + row) * N;
    float s = 0.0f;
    for (int j = lane; j < N; j += 32) s += r[j];
    #pragma unroll
    for (int off = 16; off > 0; off >>= 1) s += __shfl_xor(s, off, 32);
    if (lane == 0) dinv[(size_t)b * N + row] = rsqrtf(s);
}

// A[b][i][j] *= dinv[b][i]*dinv[b][j]
__global__ void ahat_scale(float* __restrict__ A, const float* __restrict__ dinv,
                           int N, size_t total) {
    size_t idx = (size_t)blockIdx.x * blockDim.x + threadIdx.x;
    if (idx >= total) return;
    const size_t nn = (size_t)N * N;
    const int b = (int)(idx / nn);
    const size_t rem = idx % nn;
    const int i = (int)(rem / N), j = (int)(rem % N);
    A[idx] *= dinv[(size_t)b * N + i] * dinv[(size_t)b * N + j];
}

// in-place row softmax; one wave per row, L <= 256 (L==1 degenerates to 1.0)
__global__ void softmax_rows(float* __restrict__ x, int nrows, int L) {
    const int lane = threadIdx.x & 31;
    const int row  = (int)((blockIdx.x * blockDim.x + threadIdx.x) >> 5);
    if (row >= nrows) return;
    float* r = x + (size_t)row * L;
    float m = -INFINITY;
    for (int j = lane; j < L; j += 32) m = fmaxf(m, r[j]);
    #pragma unroll
    for (int off = 16; off > 0; off >>= 1) m = fmaxf(m, __shfl_xor(m, off, 32));
    float s = 0.0f;
    for (int j = lane; j < L; j += 32) { float e = __expf(r[j] - m); s += e; r[j] = e; }
    #pragma unroll
    for (int off = 16; off > 0; off >>= 1) s += __shfl_xor(s, off, 32);
    const float inv = 1.0f / s;
    for (int j = lane; j < L; j += 32) r[j] *= inv;
}

// ---------------------------------------------------------------------------
extern "C" void kernel_launch(void* const* d_in, const int* in_sizes, int n_in,
                              void* d_out, int out_size, void* d_ws, size_t ws_size,
                              hipStream_t stream) {
    const float* X   = (const float*)d_in[0];  // [16,1024,64]
    const float* W1a = (const float*)d_in[1];  // [64,256]
    const float* W1b = (const float*)d_in[2];  // [256,128]
    const float* Ws1 = (const float*)d_in[3];  // [128,256]
    const float* W2a = (const float*)d_in[4];  // [128,256]
    const float* W2b = (const float*)d_in[5];  // [256,128]
    const float* Ws2 = (const float*)d_in[6];  // [128,64]
    const float* W3a = (const float*)d_in[7];  // [128,128]
    const float* W3b = (const float*)d_in[8];  // [128,10]
    const float* Ws3 = (const float*)d_in[9];  // [10,1]
    float* out = (float*)d_out;                // [16,10]

    // ---- workspace carving (deterministic, ~135 MB) ----
    char* ws = (char*)d_ws;
    size_t off = 0;
    auto carve = [&](size_t bytes) -> float* {
        float* p = (float*)(ws + off);
        off += (bytes + 255) & ~(size_t)255;
        return p;
    };
    float* Ah1  = carve((size_t)BATCH * 1024 * 1024 * 4); // normalized adjacency N=1024
    float* bufA = carve((size_t)BATCH * 1024 * 256 * 4);  // XW1 -> SP1 -> T1
    float* bufB = carve((size_t)BATCH * 1024 * 256 * 4);  // H1  -> s1
    float* bufC = carve((size_t)BATCH * 1024 * 128 * 4);  // H1b
    float* X2   = carve((size_t)BATCH * 1024 * 128 * 4);
    float* Ah2  = carve((size_t)BATCH * 256 * 256 * 4);
    float* XW2  = carve((size_t)BATCH * 256 * 256 * 4);
    float* H2   = carve((size_t)BATCH * 256 * 256 * 4);
    float* H2b  = carve((size_t)BATCH * 256 * 128 * 4);
    float* X3   = carve((size_t)BATCH * 256 * 128 * 4);
    float* xp2  = carve((size_t)BATCH * 256 * 128 * 4);
    float* SP2  = carve((size_t)BATCH * 256 * 64 * 4);
    float* s2   = carve((size_t)BATCH * 256 * 64 * 4);
    float* T2   = carve((size_t)BATCH * 64 * 256 * 4);
    float* xp3  = carve((size_t)BATCH * 64 * 128 * 4);
    float* Ah3  = carve((size_t)BATCH * 64 * 64 * 4);
    float* XW3  = carve((size_t)BATCH * 64 * 128 * 4);
    float* H3   = carve((size_t)BATCH * 64 * 128 * 4);
    float* H3b  = carve((size_t)BATCH * 64 * 10 * 4);
    float* X4   = carve((size_t)BATCH * 64 * 10 * 4);
    float* SP3  = carve((size_t)BATCH * 64 * 4);
    float* s3   = carve((size_t)BATCH * 64 * 4);
    float* dinv1= carve((size_t)BATCH * 1024 * 4);
    (void)in_sizes; (void)n_in; (void)out_size; (void)ws_size;

    auto gemm = [&](const float* A, const float* Bm, float* C,
                    int M, int N, int K,
                    long long sA, long long sB, long long sC,
                    int tA, int tB, int epi) {
        dim3 grid((unsigned)((N + 63) / 64), (unsigned)((M + 63) / 64), BATCH);
        if (tA)
            gemm_wmma_f16<1, 0><<<grid, dim3(128), 0, stream>>>(A, Bm, C, M, N, K, sA, sB, sC, epi);
        else if (tB)
            gemm_wmma_f16<0, 1><<<grid, dim3(128), 0, stream>>>(A, Bm, C, M, N, K, sA, sB, sC, epi);
        else
            gemm_wmma_f16<0, 0><<<grid, dim3(128), 0, stream>>>(A, Bm, C, M, N, K, sA, sB, sC, epi);
    };

    // ---------- Ah = norm(X X^T + I), N=1024 ----------
    gemm(X, X, Ah1, 1024, 1024, 64, 65536, 65536, 1048576, 0, 1, EPI_ADDI);
    rowsum_rsqrt<<<dim3(256, BATCH), 128, 0, stream>>>(Ah1, dinv1, 1024);
    {
        size_t tot = (size_t)BATCH * 1024 * 1024;
        ahat_scale<<<(unsigned)((tot + 255) / 256), 256, 0, stream>>>(Ah1, dinv1, 1024, tot);
    }

    // ---------- GCN block 1 ----------
    gemm(X,    W1a,  bufA, 1024, 256,   64,   65536,      0, 262144, 0, 0, EPI_NONE); // XW1
    gemm(Ah1,  bufA, bufB, 1024, 256, 1024, 1048576, 262144, 262144, 0, 0, EPI_RELU); // H1
    gemm(bufB, W1b,  bufC, 1024, 128,  256,  262144,      0, 131072, 0, 0, EPI_NONE); // H1b
    gemm(Ah1,  bufC, X2,   1024, 128, 1024, 1048576, 131072, 131072, 0, 0, EPI_NONE); // X2

    // ---------- HPool 1 ----------
    gemm(X2,   Ws1,  bufA, 1024, 256,  128,  131072,      0, 262144, 0, 0, EPI_NONE); // X2@Ws1
    gemm(Ah1,  bufA, bufB, 1024, 256, 1024, 1048576, 262144, 262144, 0, 0, EPI_NONE); // pre-softmax
    softmax_rows<<<dim3(4096), 128, 0, stream>>>(bufB, BATCH * 1024, 256);            // s1
    gemm(bufB, X2,   xp2,   256, 128, 1024,  262144, 131072,  32768, 1, 0, EPI_NONE); // s1^T X2
    gemm(bufB, Ah1,  bufA,  256,1024, 1024,  262144,1048576, 262144, 1, 0, EPI_NONE); // T1 = s1^T Ah1
    gemm(bufA, bufB, Ah2,   256, 256, 1024,  262144, 262144,  65536, 0, 0, EPI_NONE); // Ah2 = T1 s1

    // ---------- GCN block 2 ----------
    gemm(xp2,  W2a,  XW2,   256, 256, 128,   32768,      0,  65536, 0, 0, EPI_NONE);
    gemm(Ah2,  XW2,  H2,    256, 256, 256,   65536,  65536,  65536, 0, 0, EPI_RELU);
    gemm(H2,   W2b,  H2b,   256, 128, 256,   65536,      0,  32768, 0, 0, EPI_NONE);
    gemm(Ah2,  H2b,  X3,    256, 128, 256,   65536,  32768,  32768, 0, 0, EPI_NONE);

    // ---------- HPool 2 ----------
    gemm(X3,   Ws2,  SP2,   256,  64, 128,   32768,      0,  16384, 0, 0, EPI_NONE);
    gemm(Ah2,  SP2,  s2,    256,  64, 256,   65536,  16384,  16384, 0, 0, EPI_NONE);
    softmax_rows<<<dim3(1024), 128, 0, stream>>>(s2, BATCH * 256, 64);
    gemm(s2,   X3,   xp3,    64, 128, 256,   16384,  32768,   8192, 1, 0, EPI_NONE);  // s2^T X3
    gemm(s2,   Ah2,  T2,     64, 256, 256,   16384,  65536,  16384, 1, 0, EPI_NONE);  // s2^T Ah2
    gemm(T2,   s2,   Ah3,    64,  64, 256,   16384,  16384,   4096, 0, 0, EPI_NONE);  // Ah3

    // ---------- GCN block 3 ----------
    gemm(xp3,  W3a,  XW3,    64, 128, 128,    8192,      0,   8192, 0, 0, EPI_NONE);
    gemm(Ah3,  XW3,  H3,     64, 128,  64,    4096,   8192,   8192, 0, 0, EPI_RELU);
    gemm(H3,   W3b,  H3b,    64,  10, 128,    8192,      0,    640, 0, 0, EPI_NONE);
    gemm(Ah3,  H3b,  X4,     64,  10,  64,    4096,    640,    640, 0, 0, EPI_NONE);

    // ---------- HPool 3 -> output ----------
    gemm(X4,   Ws3,  SP3,    64,   1,  10,     640,      0,     64, 0, 0, EPI_NONE);
    gemm(Ah3,  SP3,  s3,     64,   1,  64,    4096,     64,     64, 0, 0, EPI_NONE);
    softmax_rows<<<dim3(256), 128, 0, stream>>>(s3, BATCH * 64, 1);                   // -> ones
    gemm(s3,   X4,   out,     1,  10,  64,      64,    640,     10, 1, 0, EPI_NONE);  // [16,10]
}
